// GRUDecoder_49615462203957
// MI455X (gfx1250) — matrix operand order repaired and checked by reading, not compiled, not run on hardware
//
#include <hip/hip_runtime.h>
#include <hip/hip_bf16.h>
#include <math.h>

// Problem constants (fixed by the reference harness)
#define BATCH 256
#define HID   1024
#define TSTEPS 128
#define NBLK  64                        // persistent blocks = j-tiles

// LDS row stride in bf16 elements: 1024 + 8 pad (16B) so the 16 lane rows of a
// ds_load_b128 land on different bank groups (2 accesses/bank = LDS peak).
#define LDS_ROW 1032
// 48 weight rows (3 gates x 16 cols) + 256 floats for the per-step x vector.
#define SMEM_BYTES (48 * LDS_ROW * 2 + BATCH * 4)

typedef __bf16 bf16_t;
typedef __attribute__((ext_vector_type(16))) __bf16 v16bf;
typedef __attribute__((ext_vector_type(8)))  __bf16 v8bf;
typedef __attribute__((ext_vector_type(8)))  float  v8f;

// ---------------------------------------------------------------------------
// One-time prep kernels
// ---------------------------------------------------------------------------
__global__ void k_cvt_w(const float* __restrict__ w, bf16_t* __restrict__ wb, int n) {
    int i = blockIdx.x * blockDim.x + threadIdx.x;
    if (i < n) wb[i] = (bf16_t)w[i];
}

__global__ void k_init_h(const float* __restrict__ v, float* __restrict__ hf,
                         bf16_t* __restrict__ hb, unsigned* __restrict__ cnt) {
    int i = blockIdx.x * blockDim.x + threadIdx.x;   // covers BATCH*HID exactly
    float f = v[i];
    hf[i] = f;
    hb[i] = (bf16_t)f;
    if (i == 0) *cnt = 0u;                           // grid-barrier counter
}

// Build a v16bf WMMA operand from two 16-byte loads (global or LDS).
template <typename P>
static __device__ __forceinline__ v16bf ld16(P p0, P p1) {
    v8bf lo = *(const v8bf*)p0;
    v8bf hi = *(const v8bf*)p1;
    return __builtin_shufflevector(lo, hi,
        0,1,2,3,4,5,6,7,8,9,10,11,12,13,14,15);
}

static __device__ __forceinline__ v8f wmma_bf16(v16bf a, v16bf b, v8f c) {
    return __builtin_amdgcn_wmma_f32_16x16x32_bf16(false, a, false, b,
                                                   (short)0, c, false, false);
}

// ---------------------------------------------------------------------------
// Persistent GRU decoder: 64 blocks (one per 16-wide j-tile), 256 threads
// (8 waves).  Weights for the block's j-tile live in LDS for the whole
// decode.  Per step: pipelined WMMA GEMM -> gate epilogue -> partial FFN dot
// (deterministic shfl reduction) -> device-scope barrier -> fixed-order
// reduction of 64 partials gives y (=next x, kept in LDS); block 0 writes
// the output.  h and y_part are double-buffered so one barrier/step is safe.
// ---------------------------------------------------------------------------
__global__ __launch_bounds__(256)
void k_gru_persistent(bf16_t* __restrict__ hb0, bf16_t* __restrict__ hb1,
                      float*  __restrict__ hf0, float*  __restrict__ hf1,
                      const bf16_t* __restrict__ w_hh_b,   // [3H,H] bf16
                      const float* __restrict__ w_ih,      // [3H]
                      const float* __restrict__ b_ih,      // [3H]
                      const float* __restrict__ b_hh,      // [3H]
                      const float* __restrict__ w_ffn,     // [H]
                      const float* __restrict__ b_ffn,     // [1]
                      const float* __restrict__ first_input, // [1]
                      float* __restrict__ yp0,             // [NBLK][BATCH]
                      float* __restrict__ yp1,             // [NBLK][BATCH]
                      unsigned* __restrict__ cnt,          // barrier counter
                      float* __restrict__ out)             // [BATCH][TSTEPS]
{
    extern __shared__ bf16_t smem[];                 // [48][LDS_ROW] weights
    float* xlds = (float*)(smem + 48 * LDS_ROW);     // [BATCH] current input

    const int tid   = threadIdx.x;
    const int lane  = tid & 31;
    const int wid   = tid >> 5;          // 0..7
    const int jt    = blockIdx.x;        // 0..63
    const int jbase = jt * 16;

    // ---- Stage this j-tile's weights into LDS ONCE (coalesced, uniform) ---
    for (int c = tid; c < 48 * 128; c += 256) {
        const int row   = c >> 7;        // 0..47  (gate*16 + rr)
        const int chunk = c & 127;       // 0..127
        const int g     = row >> 4;
        const int rr    = row & 15;
        const v8bf* src = (const v8bf*)(w_hh_b +
            ((size_t)(g * HID + jbase + rr)) * HID + chunk * 8);
        *(v8bf*)(smem + (size_t)row * LDS_ROW + chunk * 8) = *src;
    }
    xlds[tid] = first_input[0];          // x for step 0
    __syncthreads();

    const int half = lane >> 4;          // lane group 0/1
    const int l16  = lane & 15;
    const int brow0 = wid * 32 + l16;    // A rows for tile 0
    const int brow1 = brow0 + 16;        // A rows for tile 1
    const int j     = jbase + l16;       // B col (N) this lane holds

    // B operand (ISA B 32x16 layout) from LDS: K 8h..8h+7 and 16+8h..16+8h+7 of row j.
    const bf16_t* lr = smem + (size_t)(     l16) * LDS_ROW + half * 8;
    const bf16_t* lz = smem + (size_t)(16 + l16) * LDS_ROW + half * 8;
    const bf16_t* ln = smem + (size_t)(32 + l16) * LDS_ROW + half * 8;

    // Per-lane constants, invariant across steps.
    const float bhr = b_hh[j], bhz = b_hh[HID + j], bhn = b_hh[2 * HID + j];
    const float wir = w_ih[j], wiz = w_ih[HID + j], win = w_ih[2 * HID + j];
    const float bir = b_ih[j], biz = b_ih[HID + j], bin_ = b_ih[2 * HID + j];
    const float wfj = w_ffn[j];
    const float bf0 = b_ffn[0];

#pragma unroll 1
    for (int t = 0; t < TSTEPS; ++t) {
        const bf16_t* hin_b  = (t & 1) ? hb1 : hb0;
        const float*  hin_f  = (t & 1) ? hf1 : hf0;
        bf16_t*       hout_b = (t & 1) ? hb0 : hb1;
        float*        hout_f = (t & 1) ? hf0 : hf1;
        float*        yp     = (t & 1) ? yp1 : yp0;

        // A operand (ISA 16-bit A 16x32 layout): lane half selects K block.
        const bf16_t* pa0 = hin_b + (size_t)brow0 * HID + half * 8;
        const bf16_t* pa1 = hin_b + (size_t)brow1 * HID + half * 8;

        v8f r0 = {}, r1 = {}, z0 = {}, z1 = {}, n0 = {}, n1 = {};

        // Software-pipeline stages (register-promoted under full unroll).
        v16bf A0[3], A1[3];              // A: 2 slices in flight (global)
        v16bf BR[2], BZ[2], BN[2];       // B: 1 slice in flight (LDS)

        A0[0] = ld16(pa0,      pa0 + 16);
        A1[0] = ld16(pa1,      pa1 + 16);
        A0[1] = ld16(pa0 + 32, pa0 + 48);
        A1[1] = ld16(pa1 + 32, pa1 + 48);
        BR[0] = ld16(lr, lr + 16);
        BZ[0] = ld16(lz, lz + 16);
        BN[0] = ld16(ln, ln + 16);

#pragma unroll
        for (int s = 0; s < HID / 32; ++s) {       // 32 K slices
            const int k0 = s * 32;
            if (s + 2 < HID / 32) {                // prefetch A slice s+2
                const int ka = k0 + 64;
                A0[(s + 2) % 3] = ld16(pa0 + ka, pa0 + ka + 16);
                A1[(s + 2) % 3] = ld16(pa1 + ka, pa1 + ka + 16);
            }
            if (s + 1 < HID / 32) {                // prefetch B slice s+1
                const int kb = k0 + 32;
                BR[(s + 1) & 1] = ld16(lr + kb, lr + kb + 16);
                BZ[(s + 1) & 1] = ld16(lz + kb, lz + kb + 16);
                BN[(s + 1) & 1] = ld16(ln + kb, ln + kb + 16);
            }
            const v16bf a0 = A0[s % 3], a1 = A1[s % 3];
            const v16bf br = BR[s & 1], bz = BZ[s & 1], bn = BN[s & 1];
            r0 = wmma_bf16(a0, br, r0);
            r1 = wmma_bf16(a1, br, r1);
            z0 = wmma_bf16(a0, bz, z0);
            z1 = wmma_bf16(a1, bz, z1);
            n0 = wmma_bf16(a0, bn, n0);
            n1 = wmma_bf16(a1, bn, n1);
        }

        // ---- Gate epilogue + partial FFN dot. C/D: VGPR v = (M=half*8+v).
#pragma unroll
        for (int tile = 0; tile < 2; ++tile) {
            const v8f ar = tile ? r1 : r0;
            const v8f az = tile ? z1 : z0;
            const v8f an = tile ? n1 : n0;
#pragma unroll
            for (int v = 0; v < 8; ++v) {
                const int b = wid * 32 + tile * 16 + half * 8 + v;
                const float xv = xlds[b];
                const float gr = xv * wir + bir + ar[v] + bhr;
                const float gz = xv * wiz + biz + az[v] + bhz;
                const float gin = xv * win + bin_;
                const float r = 1.0f / (1.0f + __expf(-gr));
                const float z = 1.0f / (1.0f + __expf(-gz));
                const float n = tanhf(gin + r * (an[v] + bhn));
                const float ho = hin_f[(size_t)b * HID + j];
                const float hn = (1.0f - z) * n + z * ho;
                hout_f[(size_t)b * HID + j] = hn;
                hout_b[(size_t)b * HID + j] = (bf16_t)hn;
                // Partial dot over this block's 16 columns (lanes of the
                // half-group hold j = jbase..jbase+15 for the same b).
                float py = hn * wfj;
                py += __shfl_xor(py, 1);
                py += __shfl_xor(py, 2);
                py += __shfl_xor(py, 4);
                py += __shfl_xor(py, 8);
                if (l16 == 0) yp[jt * BATCH + b] = py;
            }
        }

        // ---- Device-scope barrier (monotonic counter, zeroed per launch) --
        __threadfence();
        __syncthreads();
        if (tid == 0) {
            __hip_atomic_fetch_add(cnt, 1u, __ATOMIC_RELEASE,
                                   __HIP_MEMORY_SCOPE_AGENT);
            const unsigned target = (unsigned)NBLK * (unsigned)(t + 1);
            while (__hip_atomic_load(cnt, __ATOMIC_ACQUIRE,
                                     __HIP_MEMORY_SCOPE_AGENT) < target)
                __builtin_amdgcn_s_sleep(2);
        }
        __syncthreads();
        __threadfence();

        // ---- Fixed-order reduction of the 64 partials: y = next x --------
        float y = bf0;
        for (int q = 0; q < NBLK; ++q) y += yp[q * BATCH + tid];
        if (jt == 0) out[(size_t)tid * TSTEPS + t] = y;
        xlds[tid] = y;
        __syncthreads();
    }
}

// ---------------------------------------------------------------------------
extern "C" void kernel_launch(void* const* d_in, const int* in_sizes, int n_in,
                              void* d_out, int out_size, void* d_ws, size_t ws_size,
                              hipStream_t stream)
{
    (void)in_sizes; (void)n_in; (void)out_size; (void)ws_size;

    const float* vectors     = (const float*)d_in[0];  // [B,H]
    const float* w_ih        = (const float*)d_in[1];  // [3H,1]
    const float* w_hh        = (const float*)d_in[2];  // [3H,H]
    const float* b_ih        = (const float*)d_in[3];  // [3H]
    const float* b_hh        = (const float*)d_in[4];  // [3H]
    const float* w_ffn       = (const float*)d_in[5];  // [1,H]
    const float* b_ffn       = (const float*)d_in[6];  // [1]
    const float* first_input = (const float*)d_in[7];  // [1]
    // d_in[8] = seq_len (device int) — fixed at TSTEPS by the reference.

    char* ws = (char*)d_ws;
    size_t off = 0;
    auto alloc = [&](size_t bytes) -> void* {
        void* p = ws + off;
        off += (bytes + 255) & ~(size_t)255;
        return p;
    };

    bf16_t*   w_hh_b = (bf16_t*)alloc((size_t)3 * HID * HID * sizeof(bf16_t));
    float*    hf0    = (float*) alloc((size_t)BATCH * HID * sizeof(float));
    float*    hf1    = (float*) alloc((size_t)BATCH * HID * sizeof(float));
    bf16_t*   hb0    = (bf16_t*)alloc((size_t)BATCH * HID * sizeof(bf16_t));
    bf16_t*   hb1    = (bf16_t*)alloc((size_t)BATCH * HID * sizeof(bf16_t));
    float*    yp0    = (float*) alloc((size_t)NBLK * BATCH * sizeof(float));
    float*    yp1    = (float*) alloc((size_t)NBLK * BATCH * sizeof(float));
    unsigned* cnt    = (unsigned*)alloc(sizeof(unsigned));

    const int nW = 3 * HID * HID;
    k_cvt_w<<<(nW + 255) / 256, 256, 0, stream>>>(w_hh, w_hh_b, nW);
    k_init_h<<<(BATCH * HID) / 256, 256, 0, stream>>>(vectors, hf0, hb0, cnt);

    k_gru_persistent<<<NBLK, 256, SMEM_BYTES, stream>>>(
        hb0, hb1, hf0, hf1, w_hh_b, w_ih, b_ih, b_hh,
        w_ffn, b_ffn, first_input, yp0, yp1, cnt, (float*)d_out);
}
